// KnowledgeGraphEncoder_1022202217078
// MI455X (gfx1250) — matrix operand order repaired and checked by reading, not compiled, not run on hardware
//
#include <hip/hip_runtime.h>
#include <stdint.h>

#define NN 50000
#define EE 600000
#define DIN 14
#define DH 128
#define DO 64
#define NHEADS 4
#define EPS 1e-5f

typedef __attribute__((ext_vector_type(16))) _Float16 v16h;
typedef __attribute__((ext_vector_type(8)))  float    v8f;

// ---------------- wave32 helpers ----------------
static __device__ __forceinline__ float warp_sum32(float v) {
#pragma unroll
  for (int m = 16; m >= 1; m >>= 1) v += __shfl_xor(v, m, 32);
  return v;
}

static __device__ __forceinline__ void atomicMaxF(float* addr, float v) {
  if (v >= 0.0f) atomicMax((int*)addr, __float_as_int(v));
  else           atomicMin((unsigned int*)addr, __float_as_uint(v));
}

// ---------------- WMMA GEMM ----------------
// out[16*bx + m, 16*(wave + 8*by) + n] = sum_k A1[m,k]*sc1(m)*W1[k,n]
//                                        (+ sum_k A2[m,k]*W2[k,n]) (+bias) (relu)
// A row-major with lda == KK; W row-major with ldw == NCOLS.
// 256 threads = 8 waves; wave w owns 16 output columns. No divergence -> EXEC all 1s.
template <int KK, int NCOLS, bool HAS_A2, bool HAS_BIAS, bool RELU, bool SCALE1>
__global__ __launch_bounds__(256) void gemm_wmma(
    const float* __restrict__ A1, const float* __restrict__ cnt,
    const float* __restrict__ W1,
    const float* __restrict__ A2, const float* __restrict__ W2,
    const float* __restrict__ bias, float* __restrict__ out) {
  const int lane = threadIdx.x & 31;
  const int wave = threadIdx.x >> 5;
  const int row_base = blockIdx.x * 16;
  const int col_base = 16 * (wave + 8 * blockIdx.y);
  const int half = lane >> 4;           // 0: lanes 0-15, 1: lanes 16-31
  const int mrow = row_base + (lane & 15);

  float sc = 1.0f;
  if (SCALE1) sc = 1.0f / fmaxf(cnt[mrow], 1.0f);

  v8f c = {};
  constexpr int KPAD = (KK + 31) & ~31;

  // ---- A1 * W1 ----
  {
    const float* ap = A1 + (size_t)mrow * KK;
#pragma unroll
    for (int k0 = 0; k0 < KPAD; k0 += 32) {
      v16h a, b;
#pragma unroll
      for (int i = 0; i < 8; ++i) {
        const int ka = k0 + half * 8 + i;          // elements 0..7
        a[i] = (_Float16)((ka < KK) ? ap[ka] * sc : 0.0f);
        const int kb = k0 + 16 + half * 8 + i;     // elements 8..15
        a[i + 8] = (_Float16)((kb < KK) ? ap[kb] * sc : 0.0f);
      }
      const int ncol = col_base + (lane & 15);     // B: lane holds column n, k in table order
      const float* bp = W1 + ncol;
#pragma unroll
      for (int i = 0; i < 16; ++i) {
        const int kr = k0 + ((i < 8) ? half * 8 + i : 16 + half * 8 + (i - 8));
        b[i] = (_Float16)((kr < KK) ? bp[(size_t)kr * NCOLS] : 0.0f);
      }
      c = __builtin_amdgcn_wmma_f32_16x16x32_f16(false, a, false, b,
                                                 (short)0, c, false, false);
    }
  }
  // ---- A2 * W2 (no scaling) ----
  if (HAS_A2) {
    const float* ap = A2 + (size_t)mrow * KK;
#pragma unroll
    for (int k0 = 0; k0 < KPAD; k0 += 32) {
      v16h a, b;
#pragma unroll
      for (int i = 0; i < 8; ++i) {
        const int ka = k0 + half * 8 + i;
        a[i] = (_Float16)((ka < KK) ? ap[ka] : 0.0f);
        const int kb = k0 + 16 + half * 8 + i;
        a[i + 8] = (_Float16)((kb < KK) ? ap[kb] : 0.0f);
      }
      const int ncol = col_base + (lane & 15);
      const float* bp = W2 + ncol;
#pragma unroll
      for (int i = 0; i < 16; ++i) {
        const int kr = k0 + ((i < 8) ? half * 8 + i : 16 + half * 8 + (i - 8));
        b[i] = (_Float16)((kr < KK) ? bp[(size_t)kr * NCOLS] : 0.0f);
      }
      c = __builtin_amdgcn_wmma_f32_16x16x32_f16(false, a, false, b,
                                                 (short)0, c, false, false);
    }
  }
  // ---- epilogue: D layout -> lanes 0-15: M=v, lanes 16-31: M=v+8 ----
  const int col = col_base + (lane & 15);
  float bv = HAS_BIAS ? bias[col] : 0.0f;
#pragma unroll
  for (int v = 0; v < 8; ++v) {
    float o = c[v] + bv;
    if (RELU) o = fmaxf(o, 0.0f);
    out[(size_t)(row_base + v + 8 * half) * NCOLS + col] = o;
  }
}

// ---------------- SAGE mean-aggregation scatter (wave per edge) ----------------
__global__ __launch_bounds__(256) void k_sage_scatter(
    const float* __restrict__ h, const int* __restrict__ src,
    const int* __restrict__ dst, float* __restrict__ s, float* __restrict__ cnt) {
  const int gid = blockIdx.x * blockDim.x + threadIdx.x;
  const int e = gid >> 5, lane = gid & 31;
  if (e >= EE) return;
  const int sn = src[e], dn = dst[e];
  const float* hp = h + (size_t)sn * DH + lane * 4;
  float* sp = s + (size_t)dn * DH + lane * 4;
#pragma unroll
  for (int i = 0; i < 4; ++i) atomicAdd(sp + i, hp[i]);
  if (lane == 0) atomicAdd(cnt + dn, 1.0f);
}

// ---------------- LN + ReLU + residual (wave per row, 128 cols) ----------------
__global__ __launch_bounds__(256) void k_ln_relu_res(
    const float* __restrict__ t, const float* __restrict__ g,
    const float* __restrict__ be, const float* __restrict__ hres,
    float* __restrict__ hout) {
  const int gid = blockIdx.x * blockDim.x + threadIdx.x;
  const int n = gid >> 5, lane = gid & 31;
  if (n >= NN) return;
  const float* tp = t + (size_t)n * DH;
  float v[4], s = 0.0f;
#pragma unroll
  for (int j = 0; j < 4; ++j) { v[j] = tp[lane + 32 * j]; s += v[j]; }
  const float mean = warp_sum32(s) * (1.0f / DH);
  float q = 0.0f;
#pragma unroll
  for (int j = 0; j < 4; ++j) { const float d = v[j] - mean; q += d * d; }
  const float inv = rsqrtf(warp_sum32(q) * (1.0f / DH) + EPS);
#pragma unroll
  for (int j = 0; j < 4; ++j) {
    const int cidx = lane + 32 * j;
    float o = (v[j] - mean) * inv * g[cidx] + be[cidx];
    hout[(size_t)n * DH + cidx] = fmaxf(o, 0.0f) + hres[(size_t)n * DH + cidx];
  }
}

// ---------------- GAT attention logits per node (wave per node) ----------------
__global__ __launch_bounds__(256) void k_att(
    const float* __restrict__ g, const float* __restrict__ att_s,
    const float* __restrict__ att_d, float* __restrict__ as_, float* __restrict__ ad_) {
  const int gid = blockIdx.x * blockDim.x + threadIdx.x;
  const int n = gid >> 5, lane = gid & 31;
  if (n >= NN) return;
  const float* gp = g + (size_t)n * (NHEADS * DO);
#pragma unroll
  for (int hd = 0; hd < NHEADS; ++hd) {
    const float x0 = gp[hd * DO + lane], x1 = gp[hd * DO + lane + 32];
    float ss = x0 * att_s[hd * DO + lane] + x1 * att_s[hd * DO + lane + 32];
    float sd = x0 * att_d[hd * DO + lane] + x1 * att_d[hd * DO + lane + 32];
    ss = warp_sum32(ss); sd = warp_sum32(sd);
    if (lane == 0) { as_[n * NHEADS + hd] = ss; ad_[n * NHEADS + hd] = sd; }
  }
}

__global__ __launch_bounds__(256) void k_fill(float* __restrict__ p, float val, int n) {
  const int i = blockIdx.x * blockDim.x + threadIdx.x;
  if (i < n) p[i] = val;
}

// edges 0..EE-1 are real; edges EE..EE+NN-1 are self-loops
static __device__ __forceinline__ void edge_pair(const int* src, const int* dst,
                                                 int e, int& sn, int& dn) {
  if (e < EE) { sn = src[e]; dn = dst[e]; }
  else        { sn = dn = e - EE; }
}

__global__ __launch_bounds__(256) void k_edge_max(
    const int* __restrict__ src, const int* __restrict__ dst,
    const float* __restrict__ as_, const float* __restrict__ ad_,
    float* __restrict__ m) {
  const int e = blockIdx.x * blockDim.x + threadIdx.x;
  if (e >= EE + NN) return;
  int sn, dn; edge_pair(src, dst, e, sn, dn);
#pragma unroll
  for (int hd = 0; hd < NHEADS; ++hd) {
    float x = as_[sn * NHEADS + hd] + ad_[dn * NHEADS + hd];
    x = (x > 0.0f) ? x : 0.2f * x;                 // leaky_relu(0.2)
    atomicMaxF(&m[dn * NHEADS + hd], x);
  }
}

__global__ __launch_bounds__(256) void k_edge_exp(
    const int* __restrict__ src, const int* __restrict__ dst,
    const float* __restrict__ as_, const float* __restrict__ ad_,
    const float* __restrict__ m, float* __restrict__ exb, float* __restrict__ den) {
  const int e = blockIdx.x * blockDim.x + threadIdx.x;
  if (e >= EE + NN) return;
  int sn, dn; edge_pair(src, dst, e, sn, dn);
#pragma unroll
  for (int hd = 0; hd < NHEADS; ++hd) {
    float x = as_[sn * NHEADS + hd] + ad_[dn * NHEADS + hd];
    x = (x > 0.0f) ? x : 0.2f * x;
    const float ex = expf(x - m[dn * NHEADS + hd]);
    exb[(size_t)e * NHEADS + hd] = ex;
    atomicAdd(&den[dn * NHEADS + hd], ex);
  }
}

// wave per edge: head-weighted message + head-mean, scatter-add 64 cols
__global__ __launch_bounds__(256) void k_edge_accum(
    const int* __restrict__ src, const int* __restrict__ dst,
    const float* __restrict__ g, const float* __restrict__ exb,
    const float* __restrict__ den, float* __restrict__ acc) {
  const int gid = blockIdx.x * blockDim.x + threadIdx.x;
  const int e = gid >> 5, lane = gid & 31;
  if (e >= EE + NN) return;
  int sn, dn; edge_pair(src, dst, e, sn, dn);
  float alpha[NHEADS];
#pragma unroll
  for (int hd = 0; hd < NHEADS; ++hd)
    alpha[hd] = exb[(size_t)e * NHEADS + hd] / den[dn * NHEADS + hd] * 0.25f;
  const float* gp = g + (size_t)sn * (NHEADS * DO);
#pragma unroll
  for (int j = 0; j < 2; ++j) {
    const int cidx = lane + 32 * j;
    float v = 0.0f;
#pragma unroll
    for (int hd = 0; hd < NHEADS; ++hd) v += alpha[hd] * gp[hd * DO + cidx];
    atomicAdd(&acc[(size_t)dn * DO + cidx], v);
  }
}

// final: LN(acc + b_g) over 64 cols (wave per row)
__global__ __launch_bounds__(256) void k_gat_post(
    const float* __restrict__ acc, const float* __restrict__ bg,
    const float* __restrict__ g3, const float* __restrict__ be3,
    float* __restrict__ out) {
  const int gid = blockIdx.x * blockDim.x + threadIdx.x;
  const int n = gid >> 5, lane = gid & 31;
  if (n >= NN) return;
  const float v0 = acc[(size_t)n * DO + lane] + bg[lane];
  const float v1 = acc[(size_t)n * DO + lane + 32] + bg[lane + 32];
  const float mean = warp_sum32(v0 + v1) * (1.0f / DO);
  const float d0 = v0 - mean, d1 = v1 - mean;
  const float inv = rsqrtf(warp_sum32(d0 * d0 + d1 * d1) * (1.0f / DO) + EPS);
  out[(size_t)n * DO + lane]      = d0 * inv * g3[lane] + be3[lane];
  out[(size_t)n * DO + lane + 32] = d1 * inv * g3[lane + 32] + be3[lane + 32];
}

// ---------------- launcher ----------------
extern "C" void kernel_launch(void* const* d_in, const int* in_sizes, int n_in,
                              void* d_out, int out_size, void* d_ws, size_t ws_size,
                              hipStream_t stream) {
  const float* x      = (const float*)d_in[0];
  const int*   ei     = (const int*)d_in[1];
  const int*   src    = ei;
  const int*   dst    = ei + EE;
  const float* W_in   = (const float*)d_in[2];
  const float* b_in   = (const float*)d_in[3];
  const float* Wl[2]  = {(const float*)d_in[4], (const float*)d_in[9]};
  const float* bb[2]  = {(const float*)d_in[5], (const float*)d_in[10]};
  const float* Wr[2]  = {(const float*)d_in[6], (const float*)d_in[11]};
  const float* gg[2]  = {(const float*)d_in[7], (const float*)d_in[12]};
  const float* bee[2] = {(const float*)d_in[8], (const float*)d_in[13]};
  const float* W_g    = (const float*)d_in[14];
  const float* attS   = (const float*)d_in[15];
  const float* attD   = (const float*)d_in[16];
  const float* b_g    = (const float*)d_in[17];
  const float* g3     = (const float*)d_in[18];
  const float* be3    = (const float*)d_in[19];

  float* ws = (float*)d_ws;
  size_t off = 0;
  auto alloc = [&](size_t nelem) { float* p = ws + off; off += nelem; return p; };
  float* h    = alloc((size_t)NN * DH);
  float* h2   = alloc((size_t)NN * DH);
  float* s    = alloc((size_t)NN * DH);
  float* tmp  = alloc((size_t)NN * DH);
  float* cnt  = alloc((size_t)NN);
  float* gbuf = alloc((size_t)NN * NHEADS * DO);
  float* as_  = alloc((size_t)NN * NHEADS);
  float* ad_  = alloc((size_t)NN * NHEADS);
  float* m    = alloc((size_t)NN * NHEADS);
  float* den  = alloc((size_t)NN * NHEADS);
  float* exb  = alloc((size_t)(EE + NN) * NHEADS);
  float* acc  = alloc((size_t)NN * DO);
  (void)ws_size; (void)in_sizes; (void)n_in; (void)out_size;

  const dim3 blk(256);
  const int rows16 = NN / 16;                       // 3125, exact

  // input projection + relu:  h = relu(x @ W_in + b_in)
  gemm_wmma<DIN, DH, false, true, true, false>
      <<<dim3(rows16, 1), blk, 0, stream>>>(x, nullptr, W_in, nullptr, nullptr, b_in, h);

  // two SAGE layers
  for (int L = 0; L < 2; ++L) {
    hipMemsetAsync(s, 0, (size_t)NN * DH * sizeof(float), stream);
    hipMemsetAsync(cnt, 0, (size_t)NN * sizeof(float), stream);
    k_sage_scatter<<<((size_t)EE * 32 + 255) / 256, blk, 0, stream>>>(h, src, dst, s, cnt);
    // tmp = (s/cnt) @ W_l + h @ W_r + b   (mean-divide fused into A-load)
    gemm_wmma<DH, DH, true, true, false, true>
        <<<dim3(rows16, 1), blk, 0, stream>>>(s, cnt, Wl[L], h, Wr[L], bb[L], tmp);
    k_ln_relu_res<<<((size_t)NN * 32 + 255) / 256, blk, 0, stream>>>(tmp, gg[L], bee[L], h, h2);
    float* t = h; h = h2; h2 = t;
  }

  // GAT: gbuf = h @ W_g  [NN, 256]
  gemm_wmma<DH, NHEADS * DO, false, false, false, false>
      <<<dim3(rows16, 2), blk, 0, stream>>>(h, nullptr, W_g, nullptr, nullptr, nullptr, gbuf);
  k_att<<<((size_t)NN * 32 + 255) / 256, blk, 0, stream>>>(gbuf, attS, attD, as_, ad_);
  k_fill<<<(NN * NHEADS + 255) / 256, blk, 0, stream>>>(m, -__builtin_inff(), NN * NHEADS);
  hipMemsetAsync(den, 0, (size_t)NN * NHEADS * sizeof(float), stream);
  hipMemsetAsync(acc, 0, (size_t)NN * DO * sizeof(float), stream);
  k_edge_max<<<(EE + NN + 255) / 256, blk, 0, stream>>>(src, dst, as_, ad_, m);
  k_edge_exp<<<(EE + NN + 255) / 256, blk, 0, stream>>>(src, dst, as_, ad_, m, exb, den);
  k_edge_accum<<<((size_t)(EE + NN) * 32 + 255) / 256, blk, 0, stream>>>(src, dst, gbuf, exb, den, acc);
  k_gat_post<<<((size_t)NN * 32 + 255) / 256, blk, 0, stream>>>(acc, b_g, g3, be3, (float*)d_out);
}
